// TransformerEncoderLayer_17961553232559
// MI455X (gfx1250) — matrix operand order repaired and checked by reading, not compiled, not run on hardware
//
#include <hip/hip_runtime.h>
#include <cstdint>
#include <cstddef>

typedef __bf16 bf16;
typedef __attribute__((ext_vector_type(16))) __bf16 v16bf;
typedef __attribute__((ext_vector_type(8)))  float  v8f;

static constexpr int B_   = 2;
static constexpr int S_   = 2048;
static constexpr int DM_  = 1024;
static constexpr int NH_  = 16;
static constexpr int HD_  = 64;
static constexpr int DFF_ = 4096;

#if __has_builtin(__builtin_amdgcn_global_load_async_to_lds_b64)
#define ASYNC_LDS 1
#endif

#ifdef ASYNC_LDS
typedef int v2i_ __attribute__((vector_size(8)));                 // 64-bit payload
typedef __attribute__((address_space(1))) v2i_* as1_v2i;          // global
typedef __attribute__((address_space(3))) v2i_* as3_v2i;          // LDS
#endif

__device__ __forceinline__ v8f v8zero() {
  v8f z; for (int i = 0; i < 8; ++i) z[i] = 0.0f; return z;
}

// copy 8 bytes global -> LDS (async path when available)
__device__ __forceinline__ void cp8_g2l(const bf16* g, bf16* l) {
#ifdef ASYNC_LDS
  __builtin_amdgcn_global_load_async_to_lds_b64((as1_v2i)g, (as3_v2i)l, 0, 0);
#else
  *(uint2*)l = *(const uint2*)g;
#endif
}

__device__ __forceinline__ void async_drain() {
#ifdef ASYNC_LDS
#if __has_builtin(__builtin_amdgcn_s_wait_asynccnt)
  __builtin_amdgcn_s_wait_asynccnt(0);
#endif
#endif
}

// load 16 bf16 from two 16B-aligned LDS chunks
__device__ __forceinline__ v16bf ld16(const bf16* p0, const bf16* p1) {
  union { uint4 u[2]; v16bf v; } t;
  t.u[0] = *(const uint4*)p0;
  t.u[1] = *(const uint4*)p1;
  return t.v;
}

__device__ __forceinline__ v8f wmma_bf16(v16bf a, v16bf b, v8f c) {
  // D = A(16x32 bf16) * B(32x16 bf16) + C(16x16 f32)
  return __builtin_amdgcn_wmma_f32_16x16x32_bf16(
      false, a, false, b, (short)0, c, false, false);
}

// ---------------- fp32 -> bf16 weight cast ----------------
__global__ __launch_bounds__(256) void k_cast_bf16(const float* __restrict__ in,
                                                   bf16* __restrict__ out, int n) {
  for (int i = blockIdx.x * 256 + threadIdx.x; i < n; i += gridDim.x * 256)
    out[i] = (bf16)in[i];
}

// ---------------- LayerNorm (rows of 1024) -> bf16 ----------------
__global__ __launch_bounds__(256) void k_ln_bf16(const float* __restrict__ x,
    const float* __restrict__ g, const float* __restrict__ bta,
    bf16* __restrict__ out) {
  const int row = blockIdx.x, t = threadIdx.x;
  const float* xr = x + (size_t)row * DM_;
  float4 v = *(const float4*)(xr + t * 4);
  float s  = v.x + v.y + v.z + v.w;
  float ss = v.x * v.x + v.y * v.y + v.z * v.z + v.w * v.w;
  for (int m = 16; m >= 1; m >>= 1) {
    s  += __shfl_xor(s,  m, 32);
    ss += __shfl_xor(ss, m, 32);
  }
  __shared__ float rs[8], rq[8];
  const int wid = t >> 5, lane = t & 31;
  if (lane == 0) { rs[wid] = s; rq[wid] = ss; }
  __syncthreads();
  float tot = 0.0f, totq = 0.0f;
  for (int i = 0; i < 8; ++i) { tot += rs[i]; totq += rq[i]; }
  const float mu  = tot * (1.0f / DM_);
  const float var = totq * (1.0f / DM_) - mu * mu;
  const float r   = rsqrtf(var + 1e-5f);
  bf16* o = out + (size_t)row * DM_;
  const int c = t * 4;
  o[c + 0] = (bf16)((v.x - mu) * r * g[c + 0] + bta[c + 0]);
  o[c + 1] = (bf16)((v.y - mu) * r * g[c + 1] + bta[c + 1]);
  o[c + 2] = (bf16)((v.z - mu) * r * g[c + 2] + bta[c + 2]);
  o[c + 3] = (bf16)((v.w - mu) * r * g[c + 3] + bta[c + 3]);
}

// ---------------- RoPE + head-split: f32 [B,S,H,hd] -> bf16 [B,H,S,hd] ----------
__global__ __launch_bounds__(256) void k_rope(const float* __restrict__ t_in,
                                              bf16* __restrict__ out) {
  int idx = blockIdx.x * 256 + threadIdx.x;  // B*S*NH*32 threads
  const int j = idx & 31;
  int r = idx >> 5;
  const int h = r & 15; r >>= 4;
  const int s = r & (S_ - 1);
  const int b = r >> 11;
  const float* p = t_in + ((size_t)(b * S_ + s)) * DM_ + h * HD_;
  const float x1 = p[j], x2 = p[j + 32];
  // inv_freq[j] = 10000^(-2j/64) = exp(-j * ln(10000)/32)
  const float th = (float)s * __expf(-0.28782313663f * (float)j);
  float sn, c;
  __sincosf(th, &sn, &c);
  bf16* o = out + (((size_t)(b * NH_ + h)) * S_ + s) * HD_;
  o[j]      = (bf16)(x1 * c - x2 * sn);
  o[j + 32] = (bf16)(x2 * c + x1 * sn);
}

// ---------------- head-split cast for V ----------------
__global__ __launch_bounds__(256) void k_vsplit(const float* __restrict__ t_in,
                                                bf16* __restrict__ out) {
  const int idx = blockIdx.x * 256 + threadIdx.x;  // B*S*DM threads
  const int d = idx & (DM_ - 1);
  int r = idx >> 10;
  const int s = r & (S_ - 1);
  const int b = r >> 11;
  const int h = d >> 6, hd = d & 63;
  out[(((size_t)(b * NH_ + h)) * S_ + s) * HD_ + hd] = (bf16)t_in[idx];
}

// ---------------- generic bf16 WMMA GEMM, 128x128 tile -------------------------
// mode 1: Cf = A*B + bias            (f32 out)
// mode 2: Cb = bf16(gelu(A*B+bias))  (bf16 out)
// mode 3: Cf = A*B + bias + resid    (f32 out)
__global__ __launch_bounds__(256) void k_gemm(const bf16* __restrict__ A,
    const bf16* __restrict__ Bm, const float* __restrict__ bias,
    const float* __restrict__ resid, float* __restrict__ Cf,
    bf16* __restrict__ Cb, int M, int N, int K, int mode) {
  __shared__ __align__(16) bf16 As[128 * 40];  // [row][k] k-stride 40
  __shared__ __align__(16) bf16 Bs[128 * 40];  // transposed: [n][k]
  const int m0 = blockIdx.y * 128, n0 = blockIdx.x * 128;
  const int t = threadIdx.x, wid = t >> 5, lane = t & 31;
  const int hl = lane >> 4, l16 = lane & 15;
  const int wm = wid >> 1, wn = wid & 1;  // wave tile: 32 rows x 64 cols
  v8f acc[2][4];
  for (int i = 0; i < 2; ++i)
    for (int j = 0; j < 4; ++j) acc[i][j] = v8zero();

  for (int k0 = 0; k0 < K; k0 += 32) {
    __syncthreads();
    // stage A tile 128x32 (async global->LDS, 8B per lane)
    for (int j = 0; j < 4; ++j) {
      const int u = t + 256 * j, row = u >> 3, c4 = u & 7;
      cp8_g2l(A + (size_t)(m0 + row) * K + k0 + c4 * 4, &As[row * 40 + c4 * 4]);
    }
    // stage B tile 32x128 transposed into [n][k]
    for (int j = 0; j < 16; ++j) {
      const int u = t + 256 * j, kk = u >> 7, n = u & 127;
      Bs[n * 40 + kk] = Bm[(size_t)(k0 + kk) * N + n0 + n];
    }
    if (k0 + 32 < K) {  // global_prefetch_b8 for next K tile
      __builtin_prefetch(A + (size_t)(m0 + (t >> 1)) * K + k0 + 32 + (t & 1) * 16, 0, 0);
      __builtin_prefetch(Bm + (size_t)(k0 + 32 + (t >> 3)) * N + n0 + (t & 7) * 16, 0, 0);
    }
    async_drain();
    __syncthreads();

    v16bf af[2], bfr[4];
    for (int i = 0; i < 2; ++i) {
      // A frag: lane row = l16; elems 0..7 -> k = hl*8.. ; elems 8..15 -> k = 16+hl*8..
      const bf16* p = As + (wm * 32 + i * 16 + l16) * 40 + hl * 8;
      af[i] = ld16(p, p + 16);
    }
    for (int j = 0; j < 4; ++j) {
      // B frag: lane col = l16; elems 0..7 -> k = hl*8.. ; elems 8..15 -> k = 16+hl*8..
      const bf16* p = Bs + (wn * 64 + j * 16 + l16) * 40 + hl * 8;
      bfr[j] = ld16(p, p + 16);
    }
    for (int i = 0; i < 2; ++i)
      for (int j = 0; j < 4; ++j)
        acc[i][j] = wmma_bf16(af[i], bfr[j], acc[i][j]);
  }

  for (int i = 0; i < 2; ++i)
    for (int j = 0; j < 4; ++j) {
      const int gcol = n0 + wn * 64 + j * 16 + l16;
      const float badd = bias ? bias[gcol] : 0.0f;
      for (int r = 0; r < 8; ++r) {
        const int grow = m0 + wm * 32 + i * 16 + hl * 8 + r;
        const size_t off = (size_t)grow * N + gcol;
        float v = acc[i][j][r] + badd;
        if (mode == 2) {
          const float ge = 0.5f * v * (1.0f + erff(v * 0.70710678f));
          Cb[off] = (bf16)ge;
        } else {
          if (mode == 3) v += resid[off];
          Cf[off] = v;
        }
      }
    }
}

// ---------------- flash attention, 128 q-rows/block, 64-key chunks -------------
__global__ __launch_bounds__(256) void k_attn(const bf16* __restrict__ Q,
    const bf16* __restrict__ Kt, const bf16* __restrict__ V,
    bf16* __restrict__ ctx) {
  __shared__ __align__(16) bf16 Ks[64 * 72];      // [key][hd]
  __shared__ __align__(16) bf16 Vs[64 * 72];      // transposed: [hd][key]
  __shared__ __align__(16) bf16 Ps[8 * 16 * 72];  // per-wave P staging [row][key]
  const int bh = blockIdx.y, qt = blockIdx.x;
  const int b = bh >> 4, h = bh & 15;
  const int t = threadIdx.x, wid = t >> 5, lane = t & 31;
  const int hl = lane >> 4, l16 = lane & 15;
  const int qrow = qt * 128 + wid * 16 + l16;

  // Q A-frags (16 rows x 64 hd = 2 frags), straight from global
  const bf16* Qb = Q + ((size_t)bh * S_ + qrow) * HD_;
  v16bf aq[2];
  for (int fa = 0; fa < 2; ++fa) {
    v16bf a;
    for (int i = 0; i < 16; ++i) {
      const int k = fa * 32 + i + hl * 8 + (i >= 8 ? 8 : 0);
      a[i] = Qb[k];
    }
    aq[fa] = a;
  }

  float mr[8], lr[8];
  for (int r = 0; r < 8; ++r) { mr[r] = -1e30f; lr[r] = 0.0f; }
  v8f accO[4];
  for (int n = 0; n < 4; ++n) accO[n] = v8zero();
  bf16* Pw = Ps + wid * 16 * 72;

  for (int c = 0; c < S_ / 64; ++c) {
    __syncthreads();
    {  // stage K chunk [64][64] -> Ks[key][hd]  (async global->LDS)
      const bf16* kg = Kt + ((size_t)bh * S_ + c * 64) * HD_;
      for (int j = 0; j < 4; ++j) {
        const int u = t + 256 * j, row = u >> 4, c4 = u & 15;
        cp8_g2l(kg + u * 4, &Ks[row * 72 + c4 * 4]);
      }
    }
    {  // stage V chunk transposed -> Vs[hd][key]
      const bf16* vg = V + ((size_t)bh * S_ + c * 64) * HD_;
      for (int j = 0; j < 16; ++j) {
        const int u = t + 256 * j;
        Vs[(u & 63) * 72 + (u >> 6)] = vg[u];
      }
    }
    async_drain();
    __syncthreads();

    // S = Q * K^T  (4 key-tiles of 16, K-dim = 64 -> 2 wmma each)
    v8f sf[4];
    for (int kt = 0; kt < 4; ++kt) {
      v8f s = v8zero();
      for (int fa = 0; fa < 2; ++fa) {
        const bf16* p = Ks + (kt * 16 + l16) * 72 + fa * 32 + hl * 8;
        s = wmma_bf16(aq[fa], ld16(p, p + 16), s);
      }
      sf[kt] = s;
    }

    // online softmax (row stats live per-lane for rows hl*8 + r)
    for (int r = 0; r < 8; ++r) {
      float sv[4], cm = -1e30f;
      for (int kt = 0; kt < 4; ++kt) {
        sv[kt] = sf[kt][r] * 0.125f;  // 1/sqrt(64)
        cm = fmaxf(cm, sv[kt]);
      }
      for (int m = 8; m >= 1; m >>= 1) cm = fmaxf(cm, __shfl_xor(cm, m, 16));
      const float mn = fmaxf(mr[r], cm);
      const float al = __expf(mr[r] - mn);
      float rs = 0.0f;
      const int prow = hl * 8 + r;
      for (int kt = 0; kt < 4; ++kt) {
        const float e = __expf(sv[kt] - mn);
        rs += e;
        Pw[prow * 72 + kt * 16 + l16] = (bf16)e;
      }
      for (int m = 8; m >= 1; m >>= 1) rs += __shfl_xor(rs, m, 16);
      lr[r] = lr[r] * al + rs;
      mr[r] = mn;
      for (int n = 0; n < 4; ++n) accO[n][r] *= al;
    }
    __syncthreads();

    // O += P * V  (K-dim = 64 keys -> 2 steps; N = 64 hd -> 4 tiles)
    for (int ks = 0; ks < 2; ++ks) {
      const bf16* pp = Pw + l16 * 72 + ks * 32 + hl * 8;
      const v16bf pa = ld16(pp, pp + 16);
      for (int n = 0; n < 4; ++n) {
        const bf16* vp = Vs + (n * 16 + l16) * 72 + ks * 32 + hl * 8;
        accO[n] = wmma_bf16(pa, ld16(vp, vp + 16), accO[n]);
      }
    }
  }

  // write ctx as [B,S,H*hd] bf16
  for (int n = 0; n < 4; ++n)
    for (int r = 0; r < 8; ++r) {
      const int row = qt * 128 + wid * 16 + hl * 8 + r;
      const float o = accO[n][r] / lr[r];
      ctx[((size_t)b * S_ + row) * DM_ + h * HD_ + n * 16 + l16] = (bf16)o;
    }
}

// =============================== host ==========================================
extern "C" void kernel_launch(void* const* d_in, const int* in_sizes, int n_in,
                              void* d_out, int out_size, void* d_ws, size_t ws_size,
                              hipStream_t stream) {
  const float* x   = (const float*)d_in[0];
  const float* Wq  = (const float*)d_in[1];
  const float* bq  = (const float*)d_in[2];
  const float* Wk  = (const float*)d_in[3];
  const float* bk  = (const float*)d_in[4];
  const float* Wv  = (const float*)d_in[5];
  const float* bv  = (const float*)d_in[6];
  const float* Wo  = (const float*)d_in[7];
  const float* bo  = (const float*)d_in[8];
  const float* g1  = (const float*)d_in[9];
  const float* be1 = (const float*)d_in[10];
  const float* W1  = (const float*)d_in[11];
  const float* b1  = (const float*)d_in[12];
  const float* W2  = (const float*)d_in[13];
  const float* b2  = (const float*)d_in[14];
  const float* g2  = (const float*)d_in[15];
  const float* be2 = (const float*)d_in[16];
  float* out = (float*)d_out;

  char* w = (char*)d_ws;
  size_t off = 0;
  auto alloc = [&](size_t bytes) -> void* {
    void* p = w + off;
    off += (bytes + 255) & ~(size_t)255;
    return p;
  };
  const int BS = B_ * S_;  // 4096
  bf16*  h1  = (bf16*)alloc((size_t)BS * DM_ * 2);
  bf16*  Wqb = (bf16*)alloc((size_t)DM_ * DM_ * 2);
  bf16*  Wkb = (bf16*)alloc((size_t)DM_ * DM_ * 2);
  bf16*  Wvb = (bf16*)alloc((size_t)DM_ * DM_ * 2);
  bf16*  Wob = (bf16*)alloc((size_t)DM_ * DM_ * 2);
  bf16*  W1b = (bf16*)alloc((size_t)DM_ * DFF_ * 2);
  bf16*  W2b = (bf16*)alloc((size_t)DFF_ * DM_ * 2);
  float* tmp = (float*)alloc((size_t)BS * DM_ * 4);
  bf16*  Qb  = (bf16*)alloc((size_t)BS * DM_ * 2);
  bf16*  Kb  = (bf16*)alloc((size_t)BS * DM_ * 2);
  bf16*  Vb  = (bf16*)alloc((size_t)BS * DM_ * 2);
  bf16*  ctx = (bf16*)alloc((size_t)BS * DM_ * 2);
  float* x1  = (float*)alloc((size_t)BS * DM_ * 4);
  bf16*  h2  = (bf16*)alloc((size_t)BS * DM_ * 2);
  bf16*  a1  = (bf16*)alloc((size_t)BS * DFF_ * 2);

  // weights -> bf16
  k_cast_bf16<<<1024, 256, 0, stream>>>(Wq, Wqb, DM_ * DM_);
  k_cast_bf16<<<1024, 256, 0, stream>>>(Wk, Wkb, DM_ * DM_);
  k_cast_bf16<<<1024, 256, 0, stream>>>(Wv, Wvb, DM_ * DM_);
  k_cast_bf16<<<1024, 256, 0, stream>>>(Wo, Wob, DM_ * DM_);
  k_cast_bf16<<<2048, 256, 0, stream>>>(W1, W1b, DM_ * DFF_);
  k_cast_bf16<<<2048, 256, 0, stream>>>(W2, W2b, DFF_ * DM_);

  // LN1
  k_ln_bf16<<<BS, 256, 0, stream>>>(x, g1, be1, h1);

  const dim3 gdd(DM_ / 128, BS / 128);  // (8,32)
  // Q,K,V projections (+bias), RoPE / head split
  k_gemm<<<gdd, 256, 0, stream>>>(h1, Wqb, bq, nullptr, tmp, nullptr, BS, DM_, DM_, 1);
  k_rope<<<(BS * DM_ / 2) / 256, 256, 0, stream>>>(tmp, Qb);
  k_gemm<<<gdd, 256, 0, stream>>>(h1, Wkb, bk, nullptr, tmp, nullptr, BS, DM_, DM_, 1);
  k_rope<<<(BS * DM_ / 2) / 256, 256, 0, stream>>>(tmp, Kb);
  k_gemm<<<gdd, 256, 0, stream>>>(h1, Wvb, bv, nullptr, tmp, nullptr, BS, DM_, DM_, 1);
  k_vsplit<<<(BS * DM_) / 256, 256, 0, stream>>>(tmp, Vb);

  // flash attention
  const dim3 ga(S_ / 128, B_ * NH_);  // (16,32)
  k_attn<<<ga, 256, 0, stream>>>(Qb, Kb, Vb, ctx);

  // O-projection + residual
  k_gemm<<<gdd, 256, 0, stream>>>(ctx, Wob, bo, x, x1, nullptr, BS, DM_, DM_, 3);

  // LN2 + FFN (GELU fused into GEMM1 epilogue) + residual
  k_ln_bf16<<<BS, 256, 0, stream>>>(x1, g2, be2, h2);
  const dim3 gff(DFF_ / 128, BS / 128);  // (32,32)
  k_gemm<<<gff, 256, 0, stream>>>(h2, W1b, b1, nullptr, nullptr, a1, BS, DFF_, DM_, 2);
  k_gemm<<<gdd, 256, 0, stream>>>(a1, W2b, b2, x1, out, nullptr, BS, DM_, DFF_, 3);

  (void)in_sizes; (void)n_in; (void)out_size; (void)ws_size;
}
